// MultibandFrameAttention_46832323395639
// MI455X (gfx1250) — matrix operand order repaired and checked by reading, not compiled, not run on hardware
//
#include <hip/hip_runtime.h>

// MultibandFrameAttention for MI455X (gfx1250, wave32, WMMA).
// B=4, W=1024, C=1024, NB=8, HD=128. Outputs: o [B,W,C] then qk [B,NB,W,W].
// Workspace layout (floats): lin(4M) qc(4M) kc(4M) vc(4M) avc(4M) = 80 MB.

#define BATCH 4
#define WFR   1024
#define CH    1024
#define NBAND 8
#define HDIM  128

typedef __attribute__((ext_vector_type(16))) _Float16 v16h;
typedef __attribute__((ext_vector_type(8)))  float    v8f;

static __device__ __forceinline__ v8f wmma_f16(v16h a, v16h b, v8f c) {
  // D = A(16x32 f16) x B(32x16 f16) + C(16x16 f32)
  return __builtin_amdgcn_wmma_f32_16x16x32_f16(
      false, a, false, b, (short)0, c, false, false);
}

// A operand (16x32, MxK) from a row-major LDS tile, row stride `ld` halves.
// Layout: lane-lo (0-15, M=lane) holds K = 0..15; lane-hi K = 16..31.
static __device__ __forceinline__ v16h frag_a_rm(const _Float16* p, int row0, int ld) {
  int lane = threadIdx.x & 31;
  int r = lane & 15, hi = lane >> 4;
  const _Float16* rp = p + (row0 + r) * ld;
  v16h f;
#pragma unroll
  for (int j = 0; j < 16; ++j) {
    int k = hi * 16 + j;
    f[j] = rp[k];
  }
  return f;
}

// B operand (32x16, KxN) taken as rows of an NxK tile (computes A * B^T).
// Layout (contiguous k, same order as frag_a_rm): lane-lo (N=lane) K=0..15, lane-hi K=16..31.
static __device__ __forceinline__ v16h frag_b_nk(const _Float16* p, int row0, int ld) {
  int lane = threadIdx.x & 31;
  int r = lane & 15, hi = lane >> 4;
  const _Float16* rp = p + (row0 + r) * ld + hi * 16;
  v16h f;
#pragma unroll
  for (int j = 0; j < 16; ++j) f[j] = rp[j];
  return f;
}

// B operand from a KxN LDS tile (computes A * B), col stride ld halves.
static __device__ __forceinline__ v16h frag_b_kn(const _Float16* p, int col0, int ld) {
  int lane = threadIdx.x & 31;
  int r = lane & 15, hi = lane >> 4;
  v16h f;
#pragma unroll
  for (int j = 0; j < 16; ++j) f[j] = p[(hi * 16 + j) * ld + col0 + r];
  return f;
}

// 16 consecutive f32 from global as 4x b128.
static __device__ __forceinline__ void load16(float* r, const float* p) {
  float4 x0 = ((const float4*)p)[0];
  float4 x1 = ((const float4*)p)[1];
  float4 x2 = ((const float4*)p)[2];
  float4 x3 = ((const float4*)p)[3];
  r[0] = x0.x;  r[1] = x0.y;  r[2] = x0.z;  r[3] = x0.w;
  r[4] = x1.x;  r[5] = x1.y;  r[6] = x1.z;  r[7] = x1.w;
  r[8] = x2.x;  r[9] = x2.y;  r[10] = x2.z; r[11] = x2.w;
  r[12] = x3.x; r[13] = x3.y; r[14] = x3.z; r[15] = x3.w;
}

static __device__ __forceinline__ void stlds16(_Float16* d, const float* r) {
#pragma unroll
  for (int j = 0; j < 16; ++j) d[j] = (_Float16)r[j];
}

// One 128x128x32 accumulate step: 8 WMMAs per wave (2 M-tiles x 4 N-tiles).
static __device__ __forceinline__ void mma_step(const _Float16* sa,
                                                const _Float16* sb, int wm,
                                                int wn, int ld, v8f acc[2][4]) {
  v16h bf[4];
#pragma unroll
  for (int j = 0; j < 4; ++j) bf[j] = frag_b_nk(sb, wn + j * 16, ld);
#pragma unroll
  for (int i = 0; i < 2; ++i) {
    v16h af = frag_a_rm(sa, wm + i * 16, ld);
#pragma unroll
    for (int j = 0; j < 4; ++j) acc[i][j] = wmma_f16(af, bf[j], acc[i][j]);
  }
}

// ---------------------------------------------------------------------------
// Generic Out[M,N] = A[M,K] @ Bt[N,K]^T  (f32 in/out, f16 WMMA, LDS ping-pong)
// 128x128 block tile, BK=32, 256 threads = 8 waves, each wave 32x64.
// ---------------------------------------------------------------------------
__global__ __launch_bounds__(256)
void gemm_xbt(const float* __restrict__ A, const float* __restrict__ Bt,
              float* __restrict__ Out, int M, int N, int K) {
  __shared__ _Float16 sA[2][128 * 40];
  __shared__ _Float16 sB[2][128 * 40];
  const int LD = 40;
  int tid = threadIdx.x;
  int wave = tid >> 5, wm = (wave & 3) * 32, wn = (wave >> 2) * 64;
  int m0 = blockIdx.x * 128, n0 = blockIdx.y * 128;
  int row = tid >> 1, cb = (tid & 1) * 16; // 128 rows x 32 cols, 16 elems/thread
  int lofs = row * LD + cb;

  const float* aptr = A + (long)(m0 + row) * K + cb;
  const float* bptr = Bt + (long)(n0 + row) * K + cb;

  v8f z = {0.f, 0.f, 0.f, 0.f, 0.f, 0.f, 0.f, 0.f};
  v8f acc[2][4];
#pragma unroll
  for (int i = 0; i < 2; ++i)
#pragma unroll
    for (int j = 0; j < 4; ++j) acc[i][j] = z;

  float ra[16], rb[16];
  load16(ra, aptr);
  load16(rb, bptr);
  stlds16(&sA[0][lofs], ra);
  stlds16(&sB[0][lofs], rb);
  __syncthreads();

  int buf = 0;
  for (int k0 = 0; k0 < K; k0 += 32) {
    bool more = (k0 + 32) < K;
    if (more) {
      load16(ra, aptr + k0 + 32);
      load16(rb, bptr + k0 + 32);
      if ((k0 + 64) < K) {
        __builtin_prefetch(aptr + k0 + 64, 0, 1); // global_prefetch_b8
        __builtin_prefetch(bptr + k0 + 64, 0, 1);
      }
    }
    mma_step(sA[buf], sB[buf], wm, wn, LD, acc);
    if (more) {
      stlds16(&sA[buf ^ 1][lofs], ra);
      stlds16(&sB[buf ^ 1][lofs], rb);
    }
    __syncthreads();
    buf ^= 1;
  }

  float* ob = Out + (long)(m0 + wm) * N + n0 + wn;
  int lane = tid & 31, nn = lane & 15, mh = (lane >> 4) * 8;
#pragma unroll
  for (int i = 0; i < 2; ++i)
#pragma unroll
    for (int j = 0; j < 4; ++j)
#pragma unroll
      for (int v = 0; v < 8; ++v)
        ob[(i * 16 + mh + v) * N + j * 16 + nn] = acc[i][j][v];
}

// ---------------------------------------------------------------------------
// Depthwise conv1d over frame axis, K=3, pad=1 (cross-correlation), float4.
// in/out: [B, W, C]; cw: [C,1,3]. One block (256 thr) covers one frame.
// ---------------------------------------------------------------------------
__global__ __launch_bounds__(256)
void dwconv(const float* __restrict__ in, const float* __restrict__ cw,
            float* __restrict__ out) {
  long idx = ((long)blockIdx.x * 256 + threadIdx.x) * 4; // over B*W*C
  int c = (int)(idx & (CH - 1));
  int w = (int)((idx >> 10) & (WFR - 1)); // CH == 1024
  float4 zer = make_float4(0.f, 0.f, 0.f, 0.f);
  float4 xm = (w > 0) ? ((const float4*)(in + idx - CH))[0] : zer;
  float4 x0 = ((const float4*)(in + idx))[0];
  float4 xp = (w < WFR - 1) ? ((const float4*)(in + idx + CH))[0] : zer;
  const float* wp = cw + c * 3; // 12 contiguous weights for channels c..c+3
  float4 o;
  o.x = xm.x * wp[0] + x0.x * wp[1] + xp.x * wp[2];
  o.y = xm.y * wp[3] + x0.y * wp[4] + xp.y * wp[5];
  o.z = xm.z * wp[6] + x0.z * wp[7] + xp.z * wp[8];
  o.w = xm.w * wp[9] + x0.w * wp[10] + xp.w * wp[11];
  ((float4*)(out + idx))[0] = o;
}

// ---------------------------------------------------------------------------
// qk[b,n,i,j] = ( dot(q_i,k_j) + dot(q_j, er[:,i]) ) / 32 + prev[b,n,i,j]
// Unified 8-step ping-pong K loop: steps 0-3 = Q_i.K_j, steps 4-7 = skew term
// (A = er columns i, B = q rows j), same accumulator.  grid: (B*NB, 8, 8)
// ---------------------------------------------------------------------------
__global__ __launch_bounds__(256)
void qk_kernel(const float* __restrict__ qc, const float* __restrict__ kc,
               const float* __restrict__ er, const float* __restrict__ prev,
               float* __restrict__ qk) {
  __shared__ _Float16 sA[2][128 * 40];
  __shared__ _Float16 sB[2][128 * 40];
  const int LD = 40;
  int bn = blockIdx.x;
  int b = bn >> 3, n = bn & 7;
  int i0 = blockIdx.y * 128, j0 = blockIdx.z * 128;
  int tid = threadIdx.x;
  int wave = tid >> 5, wm = (wave & 3) * 32, wn = (wave >> 2) * 64;
  int row = tid >> 1, cb = (tid & 1) * 16;
  int lofs = row * LD + cb;

  const float* qbase = qc + (long)b * WFR * CH + n * HDIM;
  const float* qi = qbase + (long)i0 * CH; // + row*CH + k
  const float* qj = qbase + (long)j0 * CH;
  const float* kj = kc + (long)b * WFR * CH + n * HDIM + (long)j0 * CH;

  v8f z = {0.f, 0.f, 0.f, 0.f, 0.f, 0.f, 0.f, 0.f};
  v8f acc[2][4];
#pragma unroll
  for (int i = 0; i < 2; ++i)
#pragma unroll
    for (int j = 0; j < 4; ++j) acc[i][j] = z;

  auto load_step = [&](int s, float* ra, float* rb) {
    if (s < 4) {
      int k0 = s * 32;
      load16(ra, qi + (long)row * CH + k0 + cb);
      load16(rb, kj + (long)row * CH + k0 + cb);
    } else {
      int k0 = (s - 4) * 32;
#pragma unroll
      for (int j = 0; j < 16; ++j)
        ra[j] = er[(k0 + cb + j) * WFR + i0 + row]; // er columns (strided)
      load16(rb, qj + (long)row * CH + k0 + cb);
    }
  };

  float ra[16], rb[16];
  load_step(0, ra, rb);
  stlds16(&sA[0][lofs], ra);
  stlds16(&sB[0][lofs], rb);
  __syncthreads();

  int buf = 0;
  for (int s = 0; s < 8; ++s) {
    bool more = s < 7;
    if (more) load_step(s + 1, ra, rb);
    mma_step(sA[buf], sB[buf], wm, wn, LD, acc);
    if (more) {
      stlds16(&sA[buf ^ 1][lofs], ra);
      stlds16(&sB[buf ^ 1][lofs], rb);
    }
    __syncthreads();
    buf ^= 1;
  }

  // Epilogue: scale 1/sqrt(C) and add prev_qk, write qk output (i32 offsets).
  const float* pv = prev + ((long)bn * WFR + i0) * WFR + j0;
  float* op = qk + ((long)bn * WFR + i0) * WFR + j0;
  int lane = tid & 31, nn = lane & 15, mh = (lane >> 4) * 8;
#pragma unroll
  for (int i = 0; i < 2; ++i)
#pragma unroll
    for (int j = 0; j < 4; ++j)
#pragma unroll
      for (int v = 0; v < 8; ++v) {
        int off = (wm + i * 16 + mh + v) * WFR + (wn + j * 16 + nn);
        op[off] = acc[i][j][v] * 0.03125f + pv[off];
      }
}

// ---------------------------------------------------------------------------
// Fused row-softmax + AV: av[i,h] = sum_j softmax(qk[i,:])[j] * v[j,h]
// Probabilities never hit HBM: re-exponentiated 128x32 tiles into LDS,
// ping-pong buffered against the WMMA stage.  grid: (B*NB, W/128)
// Writes avc in [b, w, n*HD+h] layout for the final output GEMM.
// ---------------------------------------------------------------------------
__global__ __launch_bounds__(256)
void softmax_av(const float* __restrict__ qk, const float* __restrict__ vc,
                float* __restrict__ avc) {
  __shared__ float red[256];
  __shared__ float rowm[128];
  __shared__ float rowsum[128];
  __shared__ _Float16 sP[2][128 * 40];
  __shared__ _Float16 sV[2][32 * 136];
  const int LDP = 40, LDV = 136;
  int bn = blockIdx.x;
  int b = bn >> 3, n = bn & 7;
  int i0 = blockIdx.y * 128;
  int tid = threadIdx.x;
  const float* qrow = qk + ((long)bn * WFR + i0) * WFR;

  // Row max / sum(exp): 2 threads per row, 512 cols each, float4 loads.
  int r = tid >> 1, half = tid & 1;
  const float4* rp4 = (const float4*)(qrow + (long)r * WFR + half * 512);
  float m = -3.0e38f;
  for (int j = 0; j < 128; ++j) {
    float4 t = rp4[j];
    m = fmaxf(m, fmaxf(fmaxf(t.x, t.y), fmaxf(t.z, t.w)));
  }
  red[tid] = m;
  __syncthreads();
  float rm = fmaxf(red[2 * r], red[2 * r + 1]);
  __syncthreads();
  float s = 0.f;
  for (int j = 0; j < 128; ++j) {
    float4 t = rp4[j];
    s += __expf(t.x - rm) + __expf(t.y - rm) + __expf(t.z - rm) +
         __expf(t.w - rm);
  }
  red[tid] = s;
  __syncthreads();
  if (half == 0) {
    rowm[r] = rm;
    rowsum[r] = red[2 * r] + red[2 * r + 1];
  }
  __syncthreads();

  int wave = tid >> 5, wm = (wave & 3) * 32, wn = (wave >> 2) * 64;
  v8f z = {0.f, 0.f, 0.f, 0.f, 0.f, 0.f, 0.f, 0.f};
  v8f acc[2][4];
#pragma unroll
  for (int i = 0; i < 2; ++i)
#pragma unroll
    for (int j = 0; j < 4; ++j) acc[i][j] = z;

  int prow = tid >> 1, pcb = (tid & 1) * 16; // P tile 128x32
  int vrow = tid >> 3, vcb = (tid & 7) * 16; // V tile 32x128 (KxN)
  int pofs = prow * LDP + pcb;
  int vofs = vrow * LDV + vcb;
  const float* vbase = vc + (long)b * WFR * CH + n * HDIM;
  float mm = rowm[prow];
  float inv = 1.0f / rowsum[prow];

  float rq[16], rv[16];
  load16(rq, qrow + (long)prow * WFR + pcb);
  load16(rv, vbase + (long)vrow * CH + vcb);
#pragma unroll
  for (int j = 0; j < 16; ++j)
    sP[0][pofs + j] = (_Float16)(__expf(rq[j] - mm) * inv);
  stlds16(&sV[0][vofs], rv);
  __syncthreads();

  int buf = 0;
  for (int j0 = 0; j0 < WFR; j0 += 32) {
    bool more = (j0 + 32) < WFR;
    if (more) {
      load16(rq, qrow + (long)prow * WFR + j0 + 32 + pcb);
      load16(rv, vbase + (long)(j0 + 32 + vrow) * CH + vcb);
    }
    { // compute on current buffers
      v16h bf[4];
#pragma unroll
      for (int jj = 0; jj < 4; ++jj)
        bf[jj] = frag_b_kn(sV[buf], wn + jj * 16, LDV);
#pragma unroll
      for (int ii = 0; ii < 2; ++ii) {
        v16h af = frag_a_rm(sP[buf], wm + ii * 16, LDP);
#pragma unroll
        for (int jj = 0; jj < 4; ++jj)
          acc[ii][jj] = wmma_f16(af, bf[jj], acc[ii][jj]);
      }
    }
    if (more) {
#pragma unroll
      for (int j = 0; j < 16; ++j)
        sP[buf ^ 1][pofs + j] = (_Float16)(__expf(rq[j] - mm) * inv);
      stlds16(&sV[buf ^ 1][vofs], rv);
    }
    __syncthreads();
    buf ^= 1;
  }

  float* ob = avc + (long)b * WFR * CH + n * HDIM + (long)i0 * CH;
  int lane = tid & 31, nn = lane & 15, mh = (lane >> 4) * 8;
#pragma unroll
  for (int ii = 0; ii < 2; ++ii)
#pragma unroll
    for (int jj = 0; jj < 4; ++jj)
#pragma unroll
      for (int v = 0; v < 8; ++v)
        ob[(wm + ii * 16 + mh + v) * CH + wn + jj * 16 + nn] = acc[ii][jj][v];
}

// ---------------------------------------------------------------------------
extern "C" void kernel_launch(void* const* d_in, const int* in_sizes, int n_in,
                              void* d_out, int out_size, void* d_ws,
                              size_t ws_size, hipStream_t stream) {
  const float* x      = (const float*)d_in[0];
  const float* prevqk = (const float*)d_in[1];
  const float* wq     = (const float*)d_in[2];
  const float* wk     = (const float*)d_in[3];
  const float* wv     = (const float*)d_in[4];
  const float* wo     = (const float*)d_in[5];
  const float* convw  = (const float*)d_in[6];
  const float* er     = (const float*)d_in[7];

  float* out = (float*)d_out;
  float* o_out  = out;                          // [B,W,C]
  float* qk_out = out + (long)BATCH * WFR * CH; // [B,NB,W,W]

  const long T = (long)BATCH * WFR * CH; // 4M floats
  float* ws  = (float*)d_ws;
  float* lin = ws;         // projection scratch (pre-conv), reused x3
  float* qc  = ws + 1 * T; // dwconv(x@wq^T)
  float* kc  = ws + 2 * T;
  float* vc  = ws + 3 * T;
  float* avc = ws + 4 * T; // attention output, [b,w,c] layout

  dim3 blk(256);
  dim3 ggemm(BATCH * WFR / 128, CH / 128);
  dim3 gconv((unsigned)(T / 4 / 256));
  int M = BATCH * WFR;

  gemm_xbt<<<ggemm, blk, 0, stream>>>(x, wq, lin, M, CH, CH);
  dwconv<<<gconv, blk, 0, stream>>>(lin, convw, qc);
  gemm_xbt<<<ggemm, blk, 0, stream>>>(x, wk, lin, M, CH, CH);
  dwconv<<<gconv, blk, 0, stream>>>(lin, convw, kc);
  gemm_xbt<<<ggemm, blk, 0, stream>>>(x, wv, lin, M, CH, CH);
  dwconv<<<gconv, blk, 0, stream>>>(lin, convw, vc);

  qk_kernel<<<dim3(BATCH * NBAND, WFR / 128, WFR / 128), blk, 0, stream>>>(
      qc, kc, er, prevqk, qk_out);
  softmax_av<<<dim3(BATCH * NBAND, WFR / 128), blk, 0, stream>>>(qk_out, vc,
                                                                 avc);
  gemm_xbt<<<ggemm, blk, 0, stream>>>(avc, wo, o_out, M, CH, CH);
}
